// Block_8495445311574
// MI455X (gfx1250) — matrix operand order repaired and checked by reading, not compiled, not run on hardware
//
#include <hip/hip_runtime.h>
#include <hip/hip_bf16.h>
#include <math.h>

typedef __bf16 bf16;
typedef __attribute__((ext_vector_type(16))) __bf16 v16bf;
typedef __attribute__((ext_vector_type(8)))  __bf16 v8bf;
typedef __attribute__((ext_vector_type(8)))  float  v8f;
typedef __attribute__((ext_vector_type(4)))  int    v4i;

static constexpr int nB  = 4;
static constexpr int nT  = 1024;
static constexpr int nC  = 1024;
static constexpr int nH  = 16;
static constexpr int nHD = 64;
static constexpr int nBT = nB * nT;        // 4096 rows for all dense GEMMs
static constexpr int SLAB = 64 * 32;       // bf16 elems per B-slab buffer (4KB)

// gfx1250 async global->LDS path (probe-guarded; falls back to load+ds_store)
#if defined(__AMDGCN__) && defined(__gfx1250__) && \
    __has_builtin(__builtin_amdgcn_global_load_async_to_lds_b128) && \
    __has_builtin(__builtin_amdgcn_s_wait_asynccnt)
#define USE_ASYNC_LDS 1
#else
#define USE_ASYNC_LDS 0
#endif

#if USE_ASYNC_LDS
#define ASYNC_COPY_B128(gsrc, ldst)                                            \
    __builtin_amdgcn_global_load_async_to_lds_b128(                            \
        (__attribute__((address_space(1))) v4i*)(gsrc),                        \
        (__attribute__((address_space(3))) v4i*)(ldst), 0, 0)
#endif

// ---------------------------------------------------------------------------
// Fragment loaders (CDNA5 wave32 WMMA 16x16x32 bf16; contiguous K order, same for A and B)
// A 16x32: lane<16 -> row=lane, K = k0..k0+15
//          lane>=16 -> row=lane-16, K = k0+16..k0+31
// B 32x16: lane<16 -> col=lane, K = k0..k0+15 ; lane>=16 -> col=lane-16, K=k0+16..
// ---------------------------------------------------------------------------
__device__ __forceinline__ v16bf load_a_frag(const bf16* __restrict__ arow, int k0, int half) {
    int ka = k0 + half * 16;
    v8bf lo = *(const v8bf*)(arow + ka);
    v8bf hi = *(const v8bf*)(arow + ka + 8);
    v16bf a;
#pragma unroll
    for (int i = 0; i < 8; ++i) { a[i] = lo[i]; a[i + 8] = hi[i]; }
    return a;
}

__device__ __forceinline__ v16bf load_a_frag(const float* __restrict__ arow, int k0, int half) {
    int ka = k0 + half * 16;
    v8f lo = *(const v8f*)(arow + ka);
    v8f hi = *(const v8f*)(arow + ka + 8);
    v16bf a;
#pragma unroll
    for (int i = 0; i < 8; ++i) { a[i] = (bf16)lo[i]; a[i + 8] = (bf16)hi[i]; }
    return a;
}

__device__ __forceinline__ v16bf load_b_frag(const bf16* __restrict__ brow, int k0, int half) {
    return *(const v16bf*)(brow + k0 + half * 16);
}

__device__ __forceinline__ v8f wmma_bf16(v16bf a, v16bf b, v8f c) {
    return __builtin_amdgcn_wmma_f32_16x16x32_bf16(false, a, false, b, (short)0, c,
                                                   false, false);
}

// ---------------------------------------------------------------------------
// Pipelined K-loop shared by all big GEMMs.
// Block = 8 waves computing a 128(M) x 64(N) tile. B is staged cooperatively
// through a double-buffered LDS slab (64 cols x 32 K bf16 = 4KB, one b128
// chunk per thread); A-fragments are register double-buffered. Next-step
// global traffic is issued before the current step's 4 WMMAs.
// ---------------------------------------------------------------------------
template <typename AT>
__device__ __forceinline__ void gemm_pipe(
    const AT* __restrict__ arow,       // this wave's A row (row = tm*16 + lr)
    const bf16* __restrict__ wblk,     // W + col0*ldw  (block's 64-column group)
    size_t ldw, int K, bf16* lds, v8f acc[4])
{
    const int tid  = threadIdx.x;
    const int lane = tid & 31;
    const int half = lane >> 4;
    const int lr   = lane & 15;
    const int ccol = tid >> 2;         // 0..63 : column this thread stages
    const int cchk = tid & 3;          // 0..3  : 8-elem K-chunk within 32
    const bf16* gsrc = wblk + (size_t)ccol * ldw + cchk * 8;
    const int lslot  = ccol * 32 + cchk * 8;

#if USE_ASYNC_LDS
    ASYNC_COPY_B128(gsrc, lds + lslot);
    __builtin_amdgcn_s_wait_asynccnt(0);
#else
    *(v8bf*)(lds + lslot) = *(const v8bf*)(gsrc);
#endif
    __syncthreads();

    v16bf af = load_a_frag(arow, 0, half);
    int cur = 0;
    for (int k0 = 0; k0 < K; k0 += 32) {
        const bool more = (k0 + 32) < K;
        v16bf afn{};
#if USE_ASYNC_LDS
        if (more) {
            ASYNC_COPY_B128(gsrc + k0 + 32, lds + (cur ^ 1) * SLAB + lslot);
            afn = load_a_frag(arow, k0 + 32, half);
        }
#else
        v8bf nxt{};
        if (more) {
            nxt = *(const v8bf*)(gsrc + k0 + 32);
            afn = load_a_frag(arow, k0 + 32, half);
        }
#endif
        const bf16* bb = lds + cur * SLAB;
#pragma unroll
        for (int t = 0; t < 4; ++t) {
            v16bf bfv = *(const v16bf*)(bb + (t * 16 + lr) * 32 + half * 16);
            acc[t] = wmma_bf16(af, bfv, acc[t]);
        }
        if (more) {
#if USE_ASYNC_LDS
            __builtin_amdgcn_s_wait_asynccnt(0);
#else
            *(v8bf*)(lds + (cur ^ 1) * SLAB + lslot) = nxt;
#endif
            __syncthreads();
            cur ^= 1;
            af = afn;
        }
    }
}

// ---------------------------------------------------------------------------
// Generic dense GEMM: out[M,N] = epilogue(A[M,K](bf16) @ W[N,K](bf16)^T + bias)
// EPI: 0 = bias -> bf16 out
//      1 = bias + exact GELU -> bf16 out
//      2 = bias + residual -> f32 out
// Grid: x = (M/128) * (N/64)
// ---------------------------------------------------------------------------
template <int EPI>
__global__ __launch_bounds__(256) void gemm_kernel(
    const bf16* __restrict__ A, const bf16* __restrict__ W,
    const float* __restrict__ bias, const float* __restrict__ res,
    void* __restrict__ outp, int M, int N, int K)
{
    __shared__ __align__(64) bf16 lds[2 * SLAB];

    const int lane = threadIdx.x & 31;
    const int half = lane >> 4;
    const int lr   = lane & 15;
    const int wave = threadIdx.x >> 5;
    const int wn   = N >> 6;
    const int tn   = blockIdx.x % wn;
    const int tm   = (blockIdx.x / wn) * 8 + wave;

    const bf16* arow = A + (size_t)(tm * 16 + lr) * K;
    const bf16* wblk = W + (size_t)(tn * 64) * K;

    v8f acc[4] = {v8f{}, v8f{}, v8f{}, v8f{}};
    gemm_pipe(arow, wblk, (size_t)K, K, lds, acc);

    const int row0 = tm * 16 + half * 8;
#pragma unroll
    for (int t = 0; t < 4; ++t) {
        const int col = tn * 64 + t * 16 + lr;
        const float bv = bias[col];
#pragma unroll
        for (int r = 0; r < 8; ++r) {
            const int row = row0 + r;
            float v = acc[t][r] + bv;
            if (EPI == 1) v = 0.5f * v * (1.0f + erff(v * 0.70710678118654752f));
            if (EPI == 2) {
                ((float*)outp)[(size_t)row * N + col] = v + res[(size_t)row * N + col];
            } else {
                ((bf16*)outp)[(size_t)row * N + col] = (bf16)v;
            }
        }
    }
}

// ---------------------------------------------------------------------------
// Attention scores: S[b,h] = (Q[b,h] @ K[b,h]^T) * 1/sqrt(HD) + mask  (fp32)
// K-depth is only 64: load all fragments up front, then 8 WMMAs back-to-back.
// ---------------------------------------------------------------------------
__global__ __launch_bounds__(256) void scores_kernel(
    const bf16* __restrict__ qkvb, const float* __restrict__ mask,
    float* __restrict__ att)
{
    const int lane = threadIdx.x & 31;
    const int half = lane >> 4;
    const int lr   = lane & 15;
    const int bh   = blockIdx.y;
    const int b    = bh >> 4;
    const int h    = bh & 15;
    const int wid  = blockIdx.x * 8 + (threadIdx.x >> 5);   // [0,1024)
    const int tm   = wid >> 4;    // 64 row tiles
    const int tn   = wid & 15;    // 16 col groups (64 wide)

    const int ld = 3 * nC;
    const bf16* qbase = qkvb + (size_t)b * nT * ld + h * nHD;
    const bf16* kbase = qbase + nC;
    const bf16* arow  = qbase + (size_t)(tm * 16 + lr) * ld;
    const bf16* bbase = kbase + (size_t)(tn * 64 + lr) * ld;

    v16bf a0 = load_a_frag(arow, 0, half);
    v16bf a1 = load_a_frag(arow, 32, half);
    v16bf b0[4], b1[4];
#pragma unroll
    for (int t = 0; t < 4; ++t) {
        b0[t] = load_b_frag(bbase + (size_t)t * 16 * ld, 0, half);
        b1[t] = load_b_frag(bbase + (size_t)t * 16 * ld, 32, half);
    }

    v8f acc[4] = {v8f{}, v8f{}, v8f{}, v8f{}};
#pragma unroll
    for (int t = 0; t < 4; ++t) acc[t] = wmma_bf16(a0, b0[t], acc[t]);
#pragma unroll
    for (int t = 0; t < 4; ++t) acc[t] = wmma_bf16(a1, b1[t], acc[t]);

    float* out = att + (size_t)bh * nT * nT;
    const int row0 = tm * 16 + half * 8;
#pragma unroll
    for (int t = 0; t < 4; ++t) {
        const int col = tn * 64 + t * 16 + lr;
#pragma unroll
        for (int r = 0; r < 8; ++r) {
            const int row = row0 + r;
            out[(size_t)row * nT + col] =
                acc[t][r] * 0.125f + mask[(size_t)row * nT + col];
        }
    }
}

// ---------------------------------------------------------------------------
// y[b,h] = P[b,h](fp32, in d_out) @ V[b,h] ; V pre-transposed so B is K-contig.
// Same LDS-staged pipeline, fp32 A converted on the fly. Block tile 128x64.
// Output packed as y[b, t, h*64 + d] (bf16) -> directly the A of the O-proj.
// ---------------------------------------------------------------------------
__global__ __launch_bounds__(256) void av_kernel(
    const float* __restrict__ att, const bf16* __restrict__ vT,
    bf16* __restrict__ yb)
{
    __shared__ __align__(64) bf16 lds[2 * SLAB];

    const int lane = threadIdx.x & 31;
    const int half = lane >> 4;
    const int lr   = lane & 15;
    const int bh   = blockIdx.y;
    const int b    = bh >> 4;
    const int h    = bh & 15;
    const int tm   = blockIdx.x * 8 + (threadIdx.x >> 5);   // [0,64)

    const float* arow = att + (size_t)bh * nT * nT + (size_t)(tm * 16 + lr) * nT;
    const bf16* wblk  = vT + (size_t)bh * nHD * nT;

    v8f acc[4] = {v8f{}, v8f{}, v8f{}, v8f{}};
    gemm_pipe(arow, wblk, (size_t)nT, nT, lds, acc);

    bf16* out = yb + (size_t)b * nT * nC + h * nHD;
    const int row0 = tm * 16 + half * 8;
#pragma unroll
    for (int t = 0; t < 4; ++t) {
        const int col = t * 16 + lr;   // d within head
#pragma unroll
        for (int r = 0; r < 8; ++r) {
            out[(size_t)(row0 + r) * nC + col] = (bf16)acc[t][r];
        }
    }
}

// ---------------------------------------------------------------------------
// Small service kernels
// ---------------------------------------------------------------------------
__global__ void cvt_bf16_kernel(const float* __restrict__ src,
                                bf16* __restrict__ dst, int n)
{
    int i = blockIdx.x * 256 + threadIdx.x;
    if (i < n) dst[i] = (bf16)src[i];
}

// vT[b,h,d,t] = qkv[b, t, 2C + h*64 + d]
__global__ void vt_kernel(const bf16* __restrict__ qkvb, bf16* __restrict__ vT)
{
    size_t i = (size_t)blockIdx.x * 256 + threadIdx.x;   // B*H*HD*T = 4M
    int t = (int)(i & 1023);
    int d = (int)((i >> 10) & 63);
    int h = (int)((i >> 16) & 15);
    int b = (int)(i >> 20);
    vT[i] = qkvb[((size_t)b * nT + t) * (3 * nC) + 2 * nC + h * nHD + d];
}

__device__ __forceinline__ float block_red_sum(float v, float* red) {
    int tid = threadIdx.x;
    red[tid] = v; __syncthreads();
    for (int s = 128; s > 0; s >>= 1) {
        if (tid < s) red[tid] += red[tid + s];
        __syncthreads();
    }
    float r = red[0]; __syncthreads();
    return r;
}

__device__ __forceinline__ float block_red_max(float v, float* red) {
    int tid = threadIdx.x;
    red[tid] = v; __syncthreads();
    for (int s = 128; s > 0; s >>= 1) {
        if (tid < s) red[tid] = fmaxf(red[tid], red[tid + s]);
        __syncthreads();
    }
    float r = red[0]; __syncthreads();
    return r;
}

// LayerNorm over C=1024, one block per row, output bf16
__global__ __launch_bounds__(256) void ln_kernel(
    const float* __restrict__ x, const float* __restrict__ w,
    const float* __restrict__ b, bf16* __restrict__ out)
{
    __shared__ float red[256];
    const int tid = threadIdx.x;
    const float* xr = x + (size_t)blockIdx.x * nC;

    float v[4];
#pragma unroll
    for (int j = 0; j < 4; ++j) v[j] = xr[tid + j * 256];

    float s = v[0] + v[1] + v[2] + v[3];
    const float mu = block_red_sum(s, red) * (1.0f / nC);

    float q = 0.f;
#pragma unroll
    for (int j = 0; j < 4; ++j) { float d = v[j] - mu; q += d * d; }
    const float var = block_red_sum(q, red) * (1.0f / nC);
    const float rs = rsqrtf(var + 1e-5f);

    bf16* orow = out + (size_t)blockIdx.x * nC;
#pragma unroll
    for (int j = 0; j < 4; ++j) {
        int i = tid + j * 256;
        orow[i] = (bf16)((v[j] - mu) * rs * w[i] + b[i]);
    }
}

// In-place row softmax over T=1024 (rows of the att buffer in d_out)
__global__ __launch_bounds__(256) void softmax_kernel(float* __restrict__ att)
{
    __shared__ float red[256];
    const int tid = threadIdx.x;
    float* p = att + (size_t)blockIdx.x * nT;

    float v[4];
#pragma unroll
    for (int j = 0; j < 4; ++j) v[j] = p[tid + j * 256];

    float m = fmaxf(fmaxf(v[0], v[1]), fmaxf(v[2], v[3]));
    m = block_red_max(m, red);

    float s = 0.f;
#pragma unroll
    for (int j = 0; j < 4; ++j) { v[j] = __expf(v[j] - m); s += v[j]; }
    s = block_red_sum(s, red);
    const float inv = 1.0f / s;

#pragma unroll
    for (int j = 0; j < 4; ++j) p[tid + j * 256] = v[j] * inv;
}

// ---------------------------------------------------------------------------
// Host-side orchestration
// ---------------------------------------------------------------------------
extern "C" void kernel_launch(void* const* d_in, const int* in_sizes, int n_in,
                              void* d_out, int out_size, void* d_ws, size_t ws_size,
                              hipStream_t stream)
{
    const float* x      = (const float*)d_in[0];
    const float* mask   = (const float*)d_in[1];
    const float* ln1_w  = (const float*)d_in[2];
    const float* ln1_b  = (const float*)d_in[3];
    const float* w_qkv  = (const float*)d_in[4];
    const float* b_qkv  = (const float*)d_in[5];
    const float* w_o    = (const float*)d_in[6];
    const float* b_o    = (const float*)d_in[7];
    const float* ln2_w  = (const float*)d_in[8];
    const float* ln2_b  = (const float*)d_in[9];
    const float* w_fc   = (const float*)d_in[10];
    const float* b_fc   = (const float*)d_in[11];
    const float* w_proj = (const float*)d_in[12];
    const float* b_proj = (const float*)d_in[13];

    float* out_x  = (float*)d_out;
    float* att    = out_x + (size_t)nB * nT * nC;   // [B,H,T,T] fp32

    char* ws = (char*)d_ws;
    size_t off = 0;
    auto bump = [&](size_t bytes) -> void* {
        void* p = ws + off;
        off += (bytes + 255) & ~(size_t)255;
        return p;
    };

    bf16* wqkvb  = (bf16*)bump((size_t)3 * nC * nC * 2);
    bf16* wob    = (bf16*)bump((size_t)nC * nC * 2);
    bf16* wfcb   = (bf16*)bump((size_t)4 * nC * nC * 2);
    bf16* wprojb = (bf16*)bump((size_t)4 * nC * nC * 2);
    bf16* h1b    = (bf16*)bump((size_t)nBT * nC * 2);
    bf16* qkvb   = (bf16*)bump((size_t)nBT * 3 * nC * 2);
    bf16* vTb    = (bf16*)bump((size_t)nB * nC * nT * 2);
    bf16* yb     = (bf16*)bump((size_t)nBT * nC * 2);
    float* x1f   = (float*)bump((size_t)nBT * nC * 4);
    bf16* h2b    = (bf16*)bump((size_t)nBT * nC * 2);
    bf16* fcb    = (bf16*)bump((size_t)nBT * 4 * nC * 2);
    (void)ws_size;

    // 1) weights fp32 -> bf16
    {
        int n;
        n = 3 * nC * nC;
        cvt_bf16_kernel<<<(n + 255) / 256, 256, 0, stream>>>(w_qkv, wqkvb, n);
        n = nC * nC;
        cvt_bf16_kernel<<<(n + 255) / 256, 256, 0, stream>>>(w_o, wob, n);
        n = 4 * nC * nC;
        cvt_bf16_kernel<<<(n + 255) / 256, 256, 0, stream>>>(w_fc, wfcb, n);
        cvt_bf16_kernel<<<(n + 255) / 256, 256, 0, stream>>>(w_proj, wprojb, n);
    }

    // 2) h = ln1(x)
    ln_kernel<<<nBT, 256, 0, stream>>>(x, ln1_w, ln1_b, h1b);

    // 3) qkv = h @ w_qkv^T + b_qkv   (M=4096, N=3072, K=1024)
    gemm_kernel<0><<<32 * 48, 256, 0, stream>>>(
        h1b, wqkvb, b_qkv, nullptr, qkvb, nBT, 3 * nC, nC);

    // 4) S = Q K^T / 8 + mask  (per b,h)
    {
        dim3 grid(128, nB * nH);
        scores_kernel<<<grid, 256, 0, stream>>>(qkvb, mask, att);
    }

    // 5) softmax rows (in place, in d_out)
    softmax_kernel<<<nB * nH * nT, 256, 0, stream>>>(att);

    // 6) V transpose: vT[b,h,d,t]
    vt_kernel<<<(nB * nC * nT) / 256, 256, 0, stream>>>(qkvb, vTb);

    // 7) y = P @ V  (per b,h), packed to [B*T, C] bf16
    {
        dim3 grid(8, nB * nH);
        av_kernel<<<grid, 256, 0, stream>>>(att, vTb, yb);
    }

    // 8) x1 = x + y @ w_o^T + b_o   (M=4096, N=1024, K=1024)
    gemm_kernel<2><<<32 * 16, 256, 0, stream>>>(
        yb, wob, b_o, x, x1f, nBT, nC, nC);

    // 9) h2 = ln2(x1)
    ln_kernel<<<nBT, 256, 0, stream>>>(x1f, ln2_w, ln2_b, h2b);

    // 10) fc = gelu(h2 @ w_fc^T + b_fc)   (M=4096, N=4096, K=1024)
    gemm_kernel<1><<<32 * 64, 256, 0, stream>>>(
        h2b, wfcb, b_fc, nullptr, fcb, nBT, 4 * nC, nC);

    // 11) out = x1 + fc @ w_proj^T + b_proj   (M=4096, N=1024, K=4096)
    gemm_kernel<2><<<32 * 16, 256, 0, stream>>>(
        fcb, wprojb, b_proj, x1f, out_x, nBT, nC, 4 * nC);

    (void)in_sizes; (void)n_in; (void)out_size;
}
